// AttnDecoder_9844065042794
// MI455X (gfx1250) — matrix operand order repaired and checked by reading, not compiled, not run on hardware
//
#include <hip/hip_runtime.h>
#include <hip/hip_bf16.h>

typedef __attribute__((ext_vector_type(16))) _Float16 v16h;
typedef __attribute__((ext_vector_type(8)))  _Float16 v8h;
typedef __attribute__((ext_vector_type(8)))  float    v8f;
typedef __attribute__((ext_vector_type(4)))  int      v4i;

#define BATCH 32
#define LSEQ  2048
#define T1    2034      // conv1 output length
#define CI    512       // conv2 input channels (=2C)
#define CO    256       // conv2 output channels (=C)
#define NT    2016      // kept conv2 columns (t in [2,2018))
#define KW    15
#define NSEC  252       // sections
#define CS    2048      // C*S

#define AS1 __attribute__((address_space(1)))
#define AS3 __attribute__((address_space(3)))

#if defined(__AMDGCN__) && __has_builtin(__builtin_amdgcn_global_load_async_to_lds_b128) && __has_builtin(__builtin_amdgcn_s_wait_asynccnt)
#define HAVE_ASYNC 1
#else
#define HAVE_ASYNC 0
#endif

// ---------------- weight preprocessing ----------------

// w1e[co][aa][k] = sum_e conv1_w[co,e,k] * aa_emb[aa,e]   (512 x 26 x 15, f32)
__global__ void make_w1e(const float* __restrict__ aa_emb,
                         const float* __restrict__ conv1_w,
                         float* __restrict__ w1e) {
    int i = blockIdx.x * 256 + threadIdx.x;
    if (i >= 512 * 26 * 15) return;
    int k  = i % 15;
    int a  = (i / 15) % 26;
    int co = i / (15 * 26);
    float s = 0.f;
#pragma unroll
    for (int e = 0; e < 5; ++e)
        s += conv1_w[(co * 5 + e) * 15 + k] * aa_emb[a * 5 + e];
    w1e[i] = s;
}

// w2h[k][co][ci] = (f16) conv2_w[co][ci][k]   (15 x 256 x 512, f16)
__global__ void make_w2h(const float* __restrict__ conv2_w,
                         _Float16* __restrict__ w2h) {
    int i = blockIdx.x * 256 + threadIdx.x;
    if (i >= KW * CO * CI) return;
    int ci = i % CI;
    int co = (i / CI) % CO;
    int k  = i / (CI * CO);
    w2h[i] = (_Float16)conv2_w[(co * CI + ci) * KW + k];
}

// ---------------- embedding + conv1 + relu ----------------
// h1t[b][t][ci] f16, t-major so conv2's B-operand reads are contiguous in ci.
__global__ __launch_bounds__(256) void conv1_kernel(
    const int* __restrict__ seq,
    const float* __restrict__ w1e,
    const float* __restrict__ b1,
    _Float16* __restrict__ h1t) {
    __shared__ int s_seq[32];
    int b     = blockIdx.x >> 7;       // 32 batches
    int chunk = blockIdx.x & 127;      // 128 chunks of 16 t
    int tbase = chunk * 16;
    if (threadIdx.x < 31) {
        int idx = tbase + threadIdx.x;
        s_seq[threadIdx.x] = (idx < LSEQ) ? seq[b * LSEQ + idx] : 0;
    }
    __syncthreads();
    for (int o = threadIdx.x; o < 512 * 16; o += 256) {
        int co = o & 511;
        int tt = o >> 9;
        int t  = tbase + tt;
        if (t >= T1) continue;
        float acc = b1[co];
        const float* wrow = w1e + (size_t)co * 26 * 15;
#pragma unroll
        for (int k = 0; k < KW; ++k)
            acc += wrow[s_seq[tt + k] * 15 + k];
        acc = acc > 0.f ? acc : 0.f;
        h1t[((size_t)b * T1 + t) * CI + co] = (_Float16)acc;
    }
}

// ---------------- conv2 via WMMA (f16 -> f32 acc) + bias + relu ----------------
// helpers: operand loaders (A and B both use contiguous-16-K per lane-half)
__device__ __forceinline__ v16h loadA16(const _Float16* p) {
    v8h lo = *(const v8h*)(p);
    v8h hi = *(const v8h*)(p + 8);
    return __builtin_shufflevector(lo, hi, 0,1,2,3,4,5,6,7,8,9,10,11,12,13,14,15);
}
__device__ __forceinline__ v16h loadB16(const _Float16* p) {
    v8h lo = *(const v8h*)(p);
    v8h hi = *(const v8h*)(p + 8);
    return __builtin_shufflevector(lo, hi, 0,1,2,3,4,5,6,7,8,9,10,11,12,13,14,15);
}

// stage one 32x512 f16 A-slab (32 KB) into LDS: 2048 x b128, 8 per thread
__device__ __forceinline__ void stage_slab(const _Float16* __restrict__ src,
                                           _Float16* __restrict__ dst, int tid) {
#if HAVE_ASYNC
#pragma unroll
    for (int j = 0; j < 8; ++j) {
        int u = tid + j * 256;
        __builtin_amdgcn_global_load_async_to_lds_b128(
            (AS1 v4i*)(src + u * 8), (AS3 v4i*)(dst + u * 8), 0, 0);
    }
#else
    const v8h* s = (const v8h*)src;
    v8h* d = (v8h*)dst;
#pragma unroll
    for (int j = 0; j < 8; ++j) {
        int u = tid + j * 256;
        d[u] = s[u];
    }
#endif
}

__device__ __forceinline__ void wait_async_copies() {
#if HAVE_ASYNC
    __builtin_amdgcn_s_wait_asynccnt(0);
#endif
}

// grid: 32(b) * 8(m-tile-pair) * 8(octet) ; block = 256 thr = 8 waves
// wave w handles n-group ng = octet*8 + w, t0 = 2 + ng*32 (63 groups cover 2016)
// each wave owns a 32(M) x 32(N) C tile -> 4 independent WMMA chains
__global__ __launch_bounds__(256) void conv2_wmma(
    const _Float16* __restrict__ w2h,   // [15][256][512]
    const _Float16* __restrict__ h1t,   // [32][2034][512]
    const float* __restrict__ b2,       // [256]
    float* __restrict__ h2) {           // [32][256][2016]
    __shared__ __align__(16) _Float16 Aslab[2][32 * CI];   // 64 KB double buffer

    const int tid  = threadIdx.x;
    const int wave = tid >> 5;
    const int lane = tid & 31;

    int bid = blockIdx.x;
    int gb  = bid & 7;          // n-group octet
    int mt2 = (bid >> 3) & 7;   // pair of 16-row m-tiles
    int b   = bid >> 6;         // 0..31
    int ng  = gb * 8 + wave;    // 0..63
    bool active = (ng < 63);
    int co0 = mt2 * 32;
    int t0  = 2 + ng * 32;

    const int m     = lane & 15;           // M row (A) / N col (B,D)
    const int halfA = (lane >> 4) * 16;    // A: contiguous-16 K offset
    const int halfB = (lane >> 4) * 16;    // B: contiguous-16 K offset

    v8f c00 = {}, c01 = {}, c10 = {}, c11 = {};

    const _Float16* h1b = h1t + (size_t)b * T1 * CI;
    const _Float16* wbase = w2h + (size_t)co0 * CI;   // + k*CO*CI per k

    // prologue: stage slab for k=0 into buffer 0
    stage_slab(wbase, Aslab[0], tid);

    for (int k = 0; k < KW; ++k) {
        wait_async_copies();
        __syncthreads();
        if (k + 1 < KW)
            stage_slab(wbase + (size_t)(k + 1) * CO * CI, Aslab[(k + 1) & 1], tid);

        if (active) {
            const _Float16* cur   = Aslab[k & 1];
            const _Float16* brow0 = h1b + (size_t)(t0 + k + m) * CI;
            if (k + 1 < KW)
                __builtin_prefetch(brow0 + CI, 0, 0);   // next k's B row
#pragma unroll 2
            for (int kb = 0; kb < CI / 32; ++kb) {
                int ciA = kb * 32 + halfA;
                v16h a0 = loadA16(cur + m * CI + ciA);
                v16h a1 = loadA16(cur + (16 + m) * CI + ciA);
                int ciB = kb * 32 + halfB;
                v16h bb0 = loadB16(brow0 + ciB);
                v16h bb1 = loadB16(brow0 + 16 * CI + ciB);

                c00 = __builtin_amdgcn_wmma_f32_16x16x32_f16(
                          false, a0, false, bb0, (short)0, c00, false, false);
                c01 = __builtin_amdgcn_wmma_f32_16x16x32_f16(
                          false, a0, false, bb1, (short)0, c01, false, false);
                c10 = __builtin_amdgcn_wmma_f32_16x16x32_f16(
                          false, a1, false, bb0, (short)0, c10, false, false);
                c11 = __builtin_amdgcn_wmma_f32_16x16x32_f16(
                          false, a1, false, bb1, (short)0, c11, false, false);
            }
        }
    }

    if (active) {
        // D layout: lane n = lane&15, VGPR r holds M = r + 8*(lane>>4)
        int n   = lane & 15;
        int mh  = (lane >> 4) * 8;
        int col = ng * 32 + n;
#pragma unroll
        for (int r = 0; r < 8; ++r) {
            int mo0 = co0 + mh + r;
            int mo1 = co0 + 16 + mh + r;
            float bi0 = b2[mo0];
            float bi1 = b2[mo1];
            float v;
            float* dst0 = h2 + ((size_t)b * CO + mo0) * NT + col;
            float* dst1 = h2 + ((size_t)b * CO + mo1) * NT + col;
            v = c00[r] + bi0; dst0[0]  = v > 0.f ? v : 0.f;
            v = c01[r] + bi0; dst0[16] = v > 0.f ? v : 0.f;
            v = c10[r] + bi1; dst1[0]  = v > 0.f ? v : 0.f;
            v = c11[r] + bi1; dst1[16] = v > 0.f ? v : 0.f;
        }
    }
}

// ---------------- fused attention: wg, energies, softmax, context ----------------
// one block per batch element
__global__ __launch_bounds__(256) void attn_kernel(
    const int* __restrict__ go_id,
    const float* __restrict__ go_table,
    const float* __restrict__ attn_w,   // [256][2048]
    const float* __restrict__ attn_b,   // [256]
    const float* __restrict__ h2,       // [32][256][2016]
    float* __restrict__ out) {          // [32][2048]
    __shared__ float wg[CS];
    __shared__ float sm[256];
    __shared__ float aN[256];
    int b   = blockIdx.x;
    int tid = threadIdx.x;
    const float* go = go_table + (size_t)go_id[b] * 256;

    // wg[p] = sum_g go[g] * attn_w[g][p]
    for (int p = tid; p < CS; p += 256) {
        float s = 0.f;
        for (int g = 0; g < 256; ++g)
            s += go[g] * attn_w[(size_t)g * CS + p];
        wg[p] = s;
    }
    // cb = go . attn_b  (tree reduce)
    sm[tid] = go[tid] * attn_b[tid];
    __syncthreads();
    for (int off = 128; off > 0; off >>= 1) {
        if (tid < off) sm[tid] += sm[tid + off];
        __syncthreads();
    }
    float cb = sm[0];
    __syncthreads();

    // energies[n] = cb + sum_{c,s} wg[c*8+s] * h2[b][c][s*252+n]
    const float* h2b = h2 + (size_t)b * CO * NT;
    float e = -1e30f;
    if (tid < NSEC) {
        float acc = cb;
        for (int c = 0; c < CO; ++c) {
            const float* row  = h2b + (size_t)c * NT;
            const float* wrow = wg + c * 8;
#pragma unroll
            for (int s = 0; s < 8; ++s)
                acc += wrow[s] * row[s * NSEC + tid];
        }
        e = acc;
    }
    // softmax over 252
    sm[tid] = e;
    __syncthreads();
    for (int off = 128; off > 0; off >>= 1) {
        if (tid < off) { float o2 = sm[tid + off]; if (o2 > sm[tid]) sm[tid] = o2; }
        __syncthreads();
    }
    float mx = sm[0];
    __syncthreads();
    float ex = (tid < NSEC) ? __expf(e - mx) : 0.f;
    sm[tid] = ex;
    __syncthreads();
    for (int off = 128; off > 0; off >>= 1) {
        if (tid < off) sm[tid] += sm[tid + off];
        __syncthreads();
    }
    float inv = 1.f / sm[0];
    aN[tid] = ex * inv;
    __syncthreads();

    // ctx[p] = sum_n aN[n] * h2[b][p/8][(p%8)*252+n]
    for (int p = tid; p < CS; p += 256) {
        const float* row = h2b + (size_t)(p >> 3) * NT + (p & 7) * NSEC;
        float acc = 0.f;
        for (int n = 0; n < NSEC; ++n)
            acc += aN[n] * row[n];
        out[(size_t)b * CS + p] = acc;
    }
}

// ---------------- launcher ----------------

extern "C" void kernel_launch(void* const* d_in, const int* in_sizes, int n_in,
                              void* d_out, int out_size, void* d_ws, size_t ws_size,
                              hipStream_t stream) {
    const int*   seq     = (const int*)d_in[0];
    const int*   go_id   = (const int*)d_in[1];
    const float* aa_emb  = (const float*)d_in[2];
    const float* conv1_w = (const float*)d_in[3];
    const float* conv1_b = (const float*)d_in[4];
    const float* conv2_w = (const float*)d_in[5];
    const float* conv2_b = (const float*)d_in[6];
    const float* go_tab  = (const float*)d_in[7];
    const float* attn_w  = (const float*)d_in[8];
    const float* attn_b  = (const float*)d_in[9];
    float* out = (float*)d_out;

    char* ws = (char*)d_ws;
    size_t off = 0;
    auto carve = [&](size_t bytes) {
        size_t p = off;
        off += (bytes + 255) & ~(size_t)255;
        return p;
    };
    float*    w1e = (float*)   (ws + carve((size_t)512 * 26 * 15 * 4));
    _Float16* w2h = (_Float16*)(ws + carve((size_t)KW * CO * CI * 2));
    _Float16* h1t = (_Float16*)(ws + carve((size_t)BATCH * T1 * CI * 2));
    float*    h2  = (float*)   (ws + carve((size_t)BATCH * CO * NT * 4));

    make_w1e<<<(512 * 26 * 15 + 255) / 256, 256, 0, stream>>>(aa_emb, conv1_w, w1e);
    make_w2h<<<(KW * CO * CI + 255) / 256, 256, 0, stream>>>(conv2_w, w2h);
    conv1_kernel<<<BATCH * 128, 256, 0, stream>>>(seq, w1e, conv1_b, h1t);
    conv2_wmma<<<BATCH * 8 * 8, 256, 0, stream>>>(w2h, h1t, conv2_b, h2);
    attn_kernel<<<BATCH, 256, 0, stream>>>(go_id, go_tab, attn_w, attn_b, h2, out);
}
